// DualTransformerPolicy_66709432041566
// MI455X (gfx1250) — matrix operand order repaired and checked by reading, not compiled, not run on hardware
//
#include <hip/hip_runtime.h>

typedef __attribute__((ext_vector_type(16))) _Float16 v16h;
typedef __attribute__((ext_vector_type(8)))  _Float16 v8h;
typedef __attribute__((ext_vector_type(8)))  float    v8f;

// ---- CDNA5 data-mover feature detection (device pass only) ----------------
#if defined(__gfx1250__)
#  if __has_builtin(__builtin_amdgcn_tensor_load_to_lds)
#    define USE_TDM 1
#    if __has_include(<hip/amd_detail/amd_gfx1250_TDM.h>)
#      define CDNA5_TDM_6ARG 1   // amdgpu-toolchain (therock) arity
#    endif
#  endif
#  if __has_builtin(__builtin_amdgcn_global_load_async_to_lds_b128)
#    define USE_ASYNC 1
#  endif
#endif

namespace {

constexpr int B    = 512;
constexpr int N    = 64;
constexpr int OBS  = 256;
constexpr int H    = 4;
constexpr int D    = 128;
constexpr int O    = 32;   // per-head value dim
constexpr int FOUT = 16;

// f16 workspace layout (offsets in _Float16 elements; all 32B-aligned)
constexpr size_t OFF_E1 = 0;
constexpr size_t OFF_K1 = OFF_E1 + (size_t)H * D * OBS;
constexpr size_t OFF_Q1 = OFF_K1 + (size_t)H * D * D;
constexpr size_t OFF_V1 = OFF_Q1 + (size_t)H * D * D;
constexpr size_t OFF_E2 = OFF_V1 + (size_t)H * O * D;
constexpr size_t OFF_K2 = OFF_E2 + (size_t)H * D * D;
constexpr size_t OFF_Q2 = OFF_K2 + (size_t)H * D * D;
constexpr size_t OFF_V2 = OFF_Q2 + (size_t)H * D * D;
constexpr size_t OFF_F1 = OFF_V2 + (size_t)H * O * D;
constexpr size_t OFF_F2 = OFF_F1 + (size_t)64 * D;
constexpr size_t OFF_X1 = OFF_F2 + (size_t)FOUT * 64;
constexpr size_t OFF_X2 = OFF_X1 + (size_t)B * N * D;

__device__ __forceinline__ float leaky(float x) { return x > 0.0f ? x : 0.01f * x; }

__device__ __forceinline__ v8f wmma_f16(v16h a, v16h b, v8f c) {
  // v_wmma_f32_16x16x32_f16: (neg_a, A, neg_b, B, c_mod, C, reuse_a, reuse_b)
  return __builtin_amdgcn_wmma_f32_16x16x32_f16(false, a, false, b, (short)0, c,
                                                false, false);
}

// ---- async / TDM tile movers ----------------------------------------------
#if defined(USE_TDM)
typedef unsigned int v4u __attribute__((ext_vector_type(4)));
typedef int          v8i __attribute__((ext_vector_type(8)));
typedef int          v4i __attribute__((ext_vector_type(4)));

// 1-D contiguous global->LDS DMA via the Tensor Data Mover.
// D# per CDNA5 ISA: count=1, type=2 ("image"),
// data_size=2 (4-byte units), tile_dim0 = #dwords (<= 65535), 1-D tile.
__device__ __forceinline__ void tdm_load_linear(const void* gsrc, void* lds_dst,
                                                unsigned nbytes) {
  const unsigned           lds_addr = (unsigned)(uintptr_t)lds_dst; // flat->LDS offset
  const unsigned long long ga       = (unsigned long long)(uintptr_t)gsrc;
  const unsigned           nwords   = nbytes >> 2;

  v4u g0;
  g0[0] = 1u;                                                  // count=1 (user D#)
  g0[1] = lds_addr;                                            // lds_addr
  g0[2] = (unsigned)ga;                                        // global_addr[31:0]
  g0[3] = (unsigned)((ga >> 32) & 0x01FFFFFFu) | (2u << 30);   // addr[56:32] | type=2

  v8i g1;
  g1[0] = (int)(2u << 16);                         // data_size=2 -> 4-byte elems
  g1[1] = (int)((nwords & 0xFFFFu) << 16);         // tensor_dim0[15:0]
  g1[2] = (int)(((nwords >> 16) & 0xFFFFu) | (1u << 16)); // tensor_dim0[31:16], tensor_dim1=1
  g1[3] = (int)((nwords & 0xFFFFu) << 16);         // tile_dim0 = nwords
  g1[4] = 0;                                       // tile_dim1=0 (1-D), tile_dim2=0
  g1[5] = (int)nwords;                             // tensor_dim0_stride[31:0]
  g1[6] = 0;
  g1[7] = 0;

  v4i zz = {0, 0, 0, 0};
#if defined(CDNA5_TDM_6ARG)
  v8i z8 = {0, 0, 0, 0, 0, 0, 0, 0};
  __builtin_amdgcn_tensor_load_to_lds(g0, g1, zz, zz, z8, 0);
#else
  __builtin_amdgcn_tensor_load_to_lds(g0, g1, zz, zz, 0);
#endif
}
#endif  // USE_TDM

#if defined(USE_ASYNC)
typedef int ai4 __attribute__((ext_vector_type(4)));

// One 16B-per-lane async global->LDS copy (wave moves 512B per issue).
// Builtin params are int4-typed AS(1)/AS(3) pointers; generic pointers
// convert implicitly in HIP device code.
__device__ __forceinline__ void async_copy_b128(const char* g, char* l) {
  __builtin_amdgcn_global_load_async_to_lds_b128((ai4*)g, (ai4*)l, 0, 0);
}
__device__ __forceinline__ void wait_async0() {
#if __has_builtin(__builtin_amdgcn_s_wait_asynccnt)
  __builtin_amdgcn_s_wait_asynccnt(0);
#else
  asm volatile("s_wait_asynccnt 0" ::: "memory");
#endif
}
#endif  // USE_ASYNC

// Contiguous f16 tile load, TDM-preferred (used for stage-2 input tiles).
// nelem*2 must be a multiple of 4096 for the async fallback (true here).
__device__ __forceinline__ void tile_load_tdm(const _Float16* __restrict__ g,
                                              _Float16* ldst, int nelem, int tid) {
#if defined(USE_TDM)
  if (tid < 32) {  // wave 0 issues the DMA; EXEC is ignored by TENSOR ops
    tdm_load_linear(g, ldst, (unsigned)nelem * 2u);
    __builtin_amdgcn_s_wait_tensorcnt(0);
  }
#elif defined(USE_ASYNC)
  const char* gs = (const char*)g + tid * 16;
  char*       ls = (char*)ldst + tid * 16;
  const int   nb = nelem * 2;
  for (int off = 0; off < nb; off += 256 * 16) async_copy_b128(gs + off, ls + off);
  wait_async0();
#else
  for (int i = tid; i < nelem; i += 256) ldst[i] = g[i];
#endif
}

// Contiguous f16 tile load, async-preferred (used for final-MLP input tiles).
__device__ __forceinline__ void tile_load_async(const _Float16* __restrict__ g,
                                                _Float16* ldst, int nelem, int tid) {
#if defined(USE_ASYNC)
  const char* gs = (const char*)g + tid * 16;
  char*       ls = (char*)ldst + tid * 16;
  const int   nb = nelem * 2;
  for (int off = 0; off < nb; off += 256 * 16) async_copy_b128(gs + off, ls + off);
  wait_async0();
#elif defined(USE_TDM)
  if (tid < 32) {
    tdm_load_linear(g, ldst, (unsigned)nelem * 2u);
    __builtin_amdgcn_s_wait_tensorcnt(0);
  }
#else
  for (int i = tid; i < nelem; i += 256) ldst[i] = g[i];
#endif
}

// A fragment: 16x32 f16 tile from row-major src[M][ld].  (ISA 7.12.2)
__device__ __forceinline__ v16h frag_a(const _Float16* __restrict__ src, int ld,
                                       int mtile, int ktile, int lane) {
  const int m    = mtile * 16 + (lane & 15);
  const int base = (lane >> 4) * 16;
  const _Float16* p = src + m * ld + ktile * 32;
  v8h lo = *(const v8h*)(p + base);
  v8h hi = *(const v8h*)(p + base + 8);
  v16h r;
#pragma unroll
  for (int i = 0; i < 8; ++i) { r[i] = lo[i]; r[i + 8] = hi[i]; }
  return r;
}

// B fragment: 32x16 f16 tile where B[k][n] = src[n*ld + k] ("N-major" source).
__device__ __forceinline__ v16h frag_b(const _Float16* __restrict__ src, int ld,
                                       int ntile, int ktile, int lane) {
  const int n = ntile * 16 + (lane & 15);
  const int k = ktile * 32 + (lane >> 4) * 16;
  return *(const v16h*)(src + n * ld + k);
}

__global__ void cvt_f32_to_f16(const float* __restrict__ src,
                               _Float16* __restrict__ dst, int n) {
  int i = blockIdx.x * 256 + threadIdx.x;
  if (i < n) dst[i] = (_Float16)src[i];
}

// One fused attention stage for one (head, batch) pair.
// I = input feature dim (256 for stage 1, 128 for stage 2).
template <int I, bool IN_F32>
__global__ __launch_bounds__(256) void attn_stage(
    const float* __restrict__ Xf32, const _Float16* __restrict__ Xf16,
    const _Float16* __restrict__ Ew, const float* __restrict__ Eb,
    const _Float16* __restrict__ Kw, const float* __restrict__ Kb,
    const _Float16* __restrict__ Qw, const float* __restrict__ Qb,
    const _Float16* __restrict__ Vw, const float* __restrict__ Vb,
    float* __restrict__ w_out, _Float16* __restrict__ x_out)
{
  const int h    = blockIdx.x;
  const int b    = blockIdx.y;
  const int tid  = threadIdx.x;
  const int lane = tid & 31;
  const int wave = tid >> 5;

  __shared__ alignas(32) _Float16 sX [N * I];   // input (f16)
  __shared__ alignas(32) _Float16 sSE[N * D];   // shared embedding
  __shared__ alignas(32) _Float16 sKm[N * D];   // keys
  __shared__ alignas(32) _Float16 sQm[N * D];   // queries
  __shared__ alignas(32) _Float16 sVT[O * N];   // values, transposed [o][m]
  __shared__ alignas(32) float    sS [N * N];   // raw scores (f32)
  __shared__ alignas(32) _Float16 sW [N * N];   // softmax weights (f16)

  const _Float16* ew = Ew + (size_t)h * D * I;  const float* eb = Eb + h * D;
  const _Float16* kw = Kw + (size_t)h * D * D;  const float* kb = Kb + h * D;
  const _Float16* qw = Qw + (size_t)h * D * D;  const float* qb = Qb + h * D;
  const _Float16* vw = Vw + (size_t)h * O * D;  const float* vb = Vb + h * O;

  // Load input tile [64 x I] into LDS.
  if (IN_F32) {
    for (int i = tid; i < N * I; i += 256)
      sX[i] = (_Float16)Xf32[(size_t)b * N * I + i];
  } else {
    tile_load_tdm(Xf16 + (size_t)b * N * I, sX, N * I, tid);  // TDM DMA path
  }
  __syncthreads();

  // se = leaky(X @ Ew^T + eb): [64 x 128]; 32 tiles over 8 waves.
  for (int t = wave; t < 32; t += 8) {
    const int mt = t & 3, nt = t >> 2;
    v8f acc = {};
#pragma unroll
    for (int kt = 0; kt < I / 32; ++kt)
      acc = wmma_f16(frag_a(sX, I, mt, kt, lane), frag_b(ew, I, nt, kt, lane), acc);
    const int n = nt * 16 + (lane & 15);
    const float bias = eb[n];
#pragma unroll
    for (int r = 0; r < 8; ++r) {
      const int m = mt * 16 + (lane >> 4) * 8 + r;
      sSE[m * D + n] = (_Float16)leaky(acc[r] + bias);
    }
  }
  __syncthreads();

  // k = se @ Kw^T + kb ; q = se @ Qw^T + qb : [64 x 128] each (64 tiles).
  for (int t = wave; t < 64; t += 8) {
    const int tt = t & 31;
    const int mt = tt & 3, nt = tt >> 2;
    const _Float16* wgt = (t < 32) ? kw : qw;
    const float*    bsp = (t < 32) ? kb : qb;
    _Float16*       dst = (t < 32) ? sKm : sQm;
    v8f acc = {};
#pragma unroll
    for (int kt = 0; kt < D / 32; ++kt)
      acc = wmma_f16(frag_a(sSE, D, mt, kt, lane), frag_b(wgt, D, nt, kt, lane), acc);
    const int n = nt * 16 + (lane & 15);
    const float bias = bsp[n];
#pragma unroll
    for (int r = 0; r < 8; ++r) {
      const int m = mt * 16 + (lane >> 4) * 8 + r;
      dst[m * D + n] = (_Float16)(acc[r] + bias);
    }
  }

  // v = leaky(se @ Vw^T + vb): [64 x 32], stored transposed as sVT[o][m].
  for (int t = wave; t < 8; t += 8) {
    const int mt = t & 3, nt = t >> 2;
    v8f acc = {};
#pragma unroll
    for (int kt = 0; kt < D / 32; ++kt)
      acc = wmma_f16(frag_a(sSE, D, mt, kt, lane), frag_b(vw, D, nt, kt, lane), acc);
    const int n = nt * 16 + (lane & 15);
    const float bias = vb[n];
#pragma unroll
    for (int r = 0; r < 8; ++r) {
      const int m = mt * 16 + (lane >> 4) * 8 + r;
      sVT[n * N + m] = (_Float16)leaky(acc[r] + bias);
    }
  }
  __syncthreads();

  // scores = (q @ k^T) / sqrt(128): [64 x 64] in f32.
  constexpr float kScale = 0.08838834764831845f;  // 1/sqrt(128)
  for (int t = wave; t < 16; t += 8) {
    const int mt = t & 3, nt = t >> 2;
    v8f acc = {};
#pragma unroll
    for (int kt = 0; kt < D / 32; ++kt)
      acc = wmma_f16(frag_a(sQm, D, mt, kt, lane), frag_b(sKm, D, nt, kt, lane), acc);
    const int n = nt * 16 + (lane & 15);
#pragma unroll
    for (int r = 0; r < 8; ++r) {
      const int m = mt * 16 + (lane >> 4) * 8 + r;
      sS[m * N + n] = acc[r] * kScale;
    }
  }
  __syncthreads();

  // Row softmax -> w_out (global f32, [H][B][N][N]) and sW (f16).
  // Column index staggered by row to avoid LDS bank conflicts.
  if (tid < N) {
    const int n = tid;
    float mx = -3.0e38f;
    for (int i = 0; i < N; ++i) {
      const int m = (i + n) & (N - 1);
      mx = fmaxf(mx, sS[n * N + m]);
    }
    float sum = 0.0f;
    for (int i = 0; i < N; ++i) {
      const int m = (i + n) & (N - 1);
      sum += __expf(sS[n * N + m] - mx);
    }
    const float inv = 1.0f / sum;
    float* wrow = w_out + (((size_t)h * B + b) * N + n) * N;
    for (int i = 0; i < N; ++i) {
      const int m = (i + n) & (N - 1);
      const float v = __expf(sS[n * N + m] - mx) * inv;
      wrow[m]       = v;
      sW[n * N + m] = (_Float16)v;
    }
  }
  __syncthreads();

  // att = w @ v : [64 x 32] -> x_out[b][m][h*32 + o]  (f16 workspace).
  for (int t = wave; t < 8; t += 8) {
    const int mt = t & 3, nt = t >> 2;
    v8f acc = {};
#pragma unroll
    for (int kt = 0; kt < N / 32; ++kt)
      acc = wmma_f16(frag_a(sW, N, mt, kt, lane), frag_b(sVT, N, nt, kt, lane), acc);
    const int n = nt * 16 + (lane & 15);
#pragma unroll
    for (int r = 0; r < 8; ++r) {
      const int m = mt * 16 + (lane >> 4) * 8 + r;
      x_out[((size_t)b * N + m) * D + h * O + n] = (_Float16)acc[r];
    }
  }
}

// Final MLP + policy softmax, one block per batch element.
__global__ __launch_bounds__(256) void final_mlp(
    const _Float16* __restrict__ X,
    const _Float16* __restrict__ F1w, const float* __restrict__ F1b,
    const _Float16* __restrict__ F2w, const float* __restrict__ F2b,
    float* __restrict__ out)
{
  const int b    = blockIdx.x;
  const int tid  = threadIdx.x;
  const int lane = tid & 31;
  const int wave = tid >> 5;

  __shared__ alignas(32) _Float16 sX[N * D];
  __shared__ alignas(32) _Float16 sH[N * 64];
  __shared__ alignas(32) float    sL[N * FOUT];

  tile_load_async(X + (size_t)b * N * D, sX, N * D, tid);  // async-LDS path
  __syncthreads();

  // hmid = leaky(x2 @ F1^T + F1b): [64 x 64]
  for (int t = wave; t < 16; t += 8) {
    const int mt = t & 3, nt = t >> 2;
    v8f acc = {};
#pragma unroll
    for (int kt = 0; kt < D / 32; ++kt)
      acc = wmma_f16(frag_a(sX, D, mt, kt, lane), frag_b(F1w, D, nt, kt, lane), acc);
    const int n = nt * 16 + (lane & 15);
    const float bias = F1b[n];
#pragma unroll
    for (int r = 0; r < 8; ++r) {
      const int m = mt * 16 + (lane >> 4) * 8 + r;
      sH[m * 64 + n] = (_Float16)leaky(acc[r] + bias);
    }
  }
  __syncthreads();

  // logits = hmid @ F2^T + F2b: [64 x 16]
  if (wave < 4) {
    const int mt = wave;
    v8f acc = {};
#pragma unroll
    for (int kt = 0; kt < 64 / 32; ++kt)
      acc = wmma_f16(frag_a(sH, 64, mt, kt, lane), frag_b(F2w, 64, 0, kt, lane), acc);
    const int n = lane & 15;
    const float bias = F2b[n];
#pragma unroll
    for (int r = 0; r < 8; ++r) {
      const int m = mt * 16 + (lane >> 4) * 8 + r;
      sL[m * FOUT + n] = acc[r] + bias;
    }
  }
  __syncthreads();

  if (tid < N) {
    const int n = tid;
    float mx = -3.0e38f;
    for (int p = 0; p < FOUT; ++p) mx = fmaxf(mx, sL[n * FOUT + p]);
    float sum = 0.0f;
    for (int p = 0; p < FOUT; ++p) sum += __expf(sL[n * FOUT + p] - mx);
    const float inv = 1.0f / sum;
    float* orow = out + ((size_t)b * N + n) * FOUT;
    for (int p = 0; p < FOUT; ++p) orow[p] = __expf(sL[n * FOUT + p] - mx) * inv;
  }
}

} // namespace

extern "C" void kernel_launch(void* const* d_in, const int* in_sizes, int n_in,
                              void* d_out, int out_size, void* d_ws, size_t ws_size,
                              hipStream_t stream) {
  (void)in_sizes; (void)n_in; (void)out_size; (void)ws_size;

  const float* states = (const float*)d_in[0];
  const float* E1w = (const float*)d_in[1];  const float* E1b = (const float*)d_in[2];
  const float* K1w = (const float*)d_in[3];  const float* K1b = (const float*)d_in[4];
  const float* Q1w = (const float*)d_in[5];  const float* Q1b = (const float*)d_in[6];
  const float* V1w = (const float*)d_in[7];  const float* V1b = (const float*)d_in[8];
  const float* E2w = (const float*)d_in[9];  const float* E2b = (const float*)d_in[10];
  const float* K2w = (const float*)d_in[11]; const float* K2b = (const float*)d_in[12];
  const float* Q2w = (const float*)d_in[13]; const float* Q2b = (const float*)d_in[14];
  const float* V2w = (const float*)d_in[15]; const float* V2b = (const float*)d_in[16];
  const float* F1w = (const float*)d_in[17]; const float* F1b = (const float*)d_in[18];
  const float* F2w = (const float*)d_in[19]; const float* F2b = (const float*)d_in[20];

  _Float16* ws = (_Float16*)d_ws;

  auto cvt = [&](const float* s, size_t off, int n) {
    cvt_f32_to_f16<<<dim3((n + 255) / 256), dim3(256), 0, stream>>>(s, ws + off, n);
  };
  cvt(E1w, OFF_E1, H * D * OBS);
  cvt(K1w, OFF_K1, H * D * D);
  cvt(Q1w, OFF_Q1, H * D * D);
  cvt(V1w, OFF_V1, H * O * D);
  cvt(E2w, OFF_E2, H * D * D);
  cvt(K2w, OFF_K2, H * D * D);
  cvt(Q2w, OFF_Q2, H * D * D);
  cvt(V2w, OFF_V2, H * O * D);
  cvt(F1w, OFF_F1, 64 * D);
  cvt(F2w, OFF_F2, FOUT * 64);

  float* pol = (float*)d_out;
  float* w1  = pol + (size_t)B * N * FOUT;
  float* w2  = w1 + (size_t)H * B * N * N;

  dim3 grid(H, B);
  attn_stage<OBS, true><<<grid, 256, 0, stream>>>(
      states, nullptr,
      ws + OFF_E1, E1b, ws + OFF_K1, K1b, ws + OFF_Q1, Q1b, ws + OFF_V1, V1b,
      w1, ws + OFF_X1);
  attn_stage<D, false><<<grid, 256, 0, stream>>>(
      nullptr, ws + OFF_X1,
      ws + OFF_E2, E2b, ws + OFF_K2, K2b, ws + OFF_Q2, Q2b, ws + OFF_V2, V2b,
      w2, ws + OFF_X2);
  final_mlp<<<dim3(B), 256, 0, stream>>>(ws + OFF_X2, ws + OFF_F1, F1b,
                                         ws + OFF_F2, F2b, pol);
}
